// myGRU_44942537785518
// MI455X (gfx1250) — matrix operand order repaired and checked by reading, not compiled, not run on hardware
//
#include <hip/hip_runtime.h>
#include <hip/hip_bf16.h>
#include <stdint.h>

#define B_   128
#define T_   1024
#define I_   256
#define H_   1024
#define O_   128
#define NSTEP (T_ - 1)
#define NBLK 64

typedef __attribute__((ext_vector_type(16))) __bf16 v16bf;
typedef __attribute__((ext_vector_type(8)))  float  v8f;

union Frag {
  v16bf v;
  uint4 q[2];
};

// round-to-nearest-even fp32 -> bf16 (bit-level, header independent)
__device__ __forceinline__ uint16_t f2bf(float x) {
  union { float f; uint32_t u; } cv; cv.f = x;
  uint32_t u = cv.u;
  if ((u & 0x7fffffffu) > 0x7f800000u) return (uint16_t)((u >> 16) | 0x0040u); // quiet NaN
  u += 0x7fffu + ((u >> 16) & 1u);
  return (uint16_t)(u >> 16);
}

// A-matrix fragment, 16-bit 16x32 (ISA 7.12.2): lane m=lane&15;
// lanes 0-15 hold K = kb+0..7 and kb+16..23 ; lanes 16-31 hold K = kb+8..15 and kb+24..31
__device__ __forceinline__ v16bf load_a_frag(const uint16_t* __restrict__ base, size_t ld,
                                             int rowbase, int kbase, int lane) {
  int m  = lane & 15;
  int kh = (lane >> 4) << 3;  // 0 or 8
  const uint16_t* p = base + (size_t)(rowbase + m) * ld + (size_t)(kbase + kh);
  Frag f;
  f.q[0] = *reinterpret_cast<const uint4*>(p);        // K = kb+kh .. +7
  f.q[1] = *reinterpret_cast<const uint4*>(p + 16);   // K = kb+kh+16 .. +23
  return f.v;
}

// B-matrix fragment, 16-bit 32x16: column n=lane&15; lanes 0-15 K=kb+0..7,16..23, lanes 16-31 K=kb+8..15,24..31.
// Weight stored row-major [N][K] so each lane's 16 K-values form two contiguous runs of 8.
__device__ __forceinline__ v16bf load_b_frag(const uint16_t* __restrict__ base, size_t ld,
                                             int nbase, int kbase, int lane) {
  int n  = lane & 15;
  int kr = (lane >> 4) << 3;  // 0 or 8
  const uint16_t* p = base + (size_t)(nbase + n) * ld + (size_t)(kbase + kr);
  Frag f;
  f.q[0] = *reinterpret_cast<const uint4*>(p);        // K = kb+kr .. +7
  f.q[1] = *reinterpret_cast<const uint4*>(p + 16);   // K = kb+kr+16 .. +23
  return f.v;
}

#define WMMA_BF16(A, Bv, C) \
  __builtin_amdgcn_wmma_f32_16x16x32_bf16(false, (A), false, (Bv), (short)0, (C), false, false)

// sense-reversing grid barrier over NBLK co-resident blocks
__device__ __forceinline__ void grid_barrier(unsigned* bar) {
  __syncthreads();
  if (threadIdx.x == 0) {
    __threadfence();
    volatile unsigned* vgen = bar + 1;
    unsigned g = *vgen;
    if (atomicAdd(bar, 1u) == NBLK - 1) {
      atomicExch(bar, 0u);
      __threadfence();
      atomicAdd(bar + 1, 1u);
    } else {
      while (*vgen == g) __builtin_amdgcn_s_sleep(1);
    }
    __threadfence();
  }
  __syncthreads();
}

__global__ void k_cvt(const float* __restrict__ in, uint16_t* __restrict__ outp, size_t n) {
  size_t i  = (size_t)blockIdx.x * blockDim.x + threadIdx.x;
  size_t st = (size_t)gridDim.x * blockDim.x;
  for (; i < n; i += st) outp[i] = f2bf(in[i]);
}

__global__ void k_init(float* __restrict__ h32, uint16_t* __restrict__ h16,
                       unsigned* __restrict__ bar) {
  size_t i  = (size_t)blockIdx.x * blockDim.x + threadIdx.x;
  size_t st = (size_t)gridDim.x * blockDim.x;
  const size_t n = (size_t)2 * B_ * H_;
  for (size_t j = i; j < n; j += st) { h32[j] = 0.f; h16[j] = 0; }
  if (i == 0) { bar[0] = 0u; bar[1] = 0u; }
}

__global__ __launch_bounds__(256)
void gru_persistent(const float* __restrict__ b_ih, const float* __restrict__ b_hh,
                    const float* __restrict__ b_out, float* __restrict__ out,
                    const uint16_t* __restrict__ Whh, const uint16_t* __restrict__ Wih,
                    const uint16_t* __restrict__ Wout, const uint16_t* __restrict__ seqb,
                    float* __restrict__ h32, uint16_t* __restrict__ h16,
                    unsigned* __restrict__ bar) {
  __shared__ float slog[16][O_];

  const int tid  = threadIdx.x;
  const int lane = tid & 31;
  const int w    = tid >> 5;          // 8 waves / block
  const int blk  = blockIdx.x;
  const int rt = blk & 3;             // 4 row tiles of 32 batch rows
  const int jt = blk >> 2;            // 16 tiles of 64 hidden units
  const int mf = w & 1;
  const int nf = w >> 1;
  const int rowbase = rt * 32 + mf * 16;
  const int jbase   = jt * 64 + nf * 16;

  const int jcol = jbase + (lane & 15);
  const float br  = b_ih[jcol]          + b_hh[jcol];
  const float bz  = b_ih[H_ + jcol]     + b_hh[H_ + jcol];
  const float bin = b_ih[2 * H_ + jcol];
  const float bhn = b_hh[2 * H_ + jcol];
  const int   ocol = (w << 4) + (lane & 15);
  const float bo   = b_out[ocol];

  for (int t = 0; t < NSTEP; ++t) {
    const uint16_t* hb  = h16 + (size_t)(t & 1) * (B_ * H_);
    uint16_t*       hbn = h16 + (size_t)((t + 1) & 1) * (B_ * H_);
    const float*    hf  = h32 + (size_t)(t & 1) * (B_ * H_);
    float*          hfn = h32 + (size_t)((t + 1) & 1) * (B_ * H_);

    // ---- phase A: gates + hidden update for a 32-row x 64-hidden slab ----
    v8f ar = {}, az = {}, ain = {}, ahn = {};
    for (int k = 0; k < H_; k += 32) {                       // recurrent GEMM, K=1024
      v16bf a  = load_a_frag(hb, H_, rowbase, k, lane);
      v16bf b0 = load_b_frag(Whh, H_, 0 * H_ + jbase, k, lane);
      v16bf b1 = load_b_frag(Whh, H_, 1 * H_ + jbase, k, lane);
      v16bf b2 = load_b_frag(Whh, H_, 2 * H_ + jbase, k, lane);
      ar  = WMMA_BF16(a, b0, ar);
      az  = WMMA_BF16(a, b1, az);
      ahn = WMMA_BF16(a, b2, ahn);                           // gh_n kept separate (r * h_n)
    }
    const uint16_t* xbase = seqb + (size_t)t * I_;           // seq bf16, [B][T][I]
    for (int k = 0; k < I_; k += 32) {                       // input GEMM, K=256
      v16bf a  = load_a_frag(xbase, (size_t)T_ * I_, rowbase, k, lane);
      v16bf b0 = load_b_frag(Wih, I_, 0 * H_ + jbase, k, lane);
      v16bf b1 = load_b_frag(Wih, I_, 1 * H_ + jbase, k, lane);
      v16bf b2 = load_b_frag(Wih, I_, 2 * H_ + jbase, k, lane);
      ar  = WMMA_BF16(a, b0, ar);                            // gi_r + gh_r fused
      az  = WMMA_BF16(a, b1, az);                            // gi_z + gh_z fused
      ain = WMMA_BF16(a, b2, ain);                           // gi_n separate
    }
    #pragma unroll
    for (int v = 0; v < 8; ++v) {
      int m = v + ((lane >> 4) << 3);                        // C/D layout
      size_t idx = (size_t)(rowbase + m) * H_ + jcol;
      float r = 1.f / (1.f + __expf(-(ar[v] + br)));
      float z = 1.f / (1.f + __expf(-(az[v] + bz)));
      float n = tanhf(ain[v] + bin + r * (ahn[v] + bhn));
      float hnew = (1.f - z) * n + z * hf[idx];              // fp32 master state
      hfn[idx] = hnew;
      hbn[idx] = f2bf(hnew);
    }

    grid_barrier(bar);                                        // h(t+1) fully visible

    // ---- phase B: output projection + log-softmax (blocks 0..7, 16 rows each) ----
    if (blk < 8) {
      const int rg = blk * 16;
      v8f acc = {};
      for (int k = 0; k < H_; k += 32) {
        v16bf a = load_a_frag(hbn, H_, rg, k, lane);
        v16bf b = load_b_frag(Wout, H_, w * 16, k, lane);
        acc = WMMA_BF16(a, b, acc);
      }
      #pragma unroll
      for (int v = 0; v < 8; ++v) {
        int m = v + ((lane >> 4) << 3);
        slog[m][ocol] = acc[v] + bo;
      }
      __syncthreads();
      if (tid < 16) {
        float mx = -3.402823466e38f;
        for (int c = 0; c < O_; ++c) mx = fmaxf(mx, slog[tid][c]);
        float s = 0.f;
        for (int c = 0; c < O_; ++c) s += __expf(slog[tid][c] - mx);
        float lse = mx + __logf(s);
        float* op = out + ((size_t)(rg + tid) * NSTEP + t) * O_;
        for (int c = 0; c < O_; ++c) op[c] = slog[tid][c] - lse;
      }
      __syncthreads();
    }
  }
}

extern "C" void kernel_launch(void* const* d_in, const int* in_sizes, int n_in,
                              void* d_out, int out_size, void* d_ws, size_t ws_size,
                              hipStream_t stream) {
  const float* seq   = (const float*)d_in[0];   // [B,T,I]
  const float* W_ih  = (const float*)d_in[1];   // [3H,I]
  const float* W_hh  = (const float*)d_in[2];   // [3H,H]
  const float* b_ih  = (const float*)d_in[3];   // [3H]
  const float* b_hh  = (const float*)d_in[4];   // [3H]
  const float* W_out = (const float*)d_in[5];   // [O,H]
  const float* b_out = (const float*)d_in[6];   // [O]
  float* out = (float*)d_out;                   // [B,T-1,O]

  char* p = (char*)d_ws;
  uint16_t* Whh_b  = (uint16_t*)p; p += (size_t)3 * H_ * H_ * 2;
  uint16_t* Wih_b  = (uint16_t*)p; p += (size_t)3 * H_ * I_ * 2;
  uint16_t* Wout_b = (uint16_t*)p; p += (size_t)O_ * H_ * 2;
  uint16_t* seqb   = (uint16_t*)p; p += (size_t)B_ * T_ * I_ * 2;
  float*    h32    = (float*)p;    p += (size_t)2 * B_ * H_ * 4;
  uint16_t* h16    = (uint16_t*)p; p += (size_t)2 * B_ * H_ * 2;
  unsigned* bar    = (unsigned*)p;

  k_cvt<<<2048, 256, 0, stream>>>(W_hh,  Whh_b,  (size_t)3 * H_ * H_);
  k_cvt<<<1024, 256, 0, stream>>>(W_ih,  Wih_b,  (size_t)3 * H_ * I_);
  k_cvt<<<256,  256, 0, stream>>>(W_out, Wout_b, (size_t)O_ * H_);
  k_cvt<<<8192, 256, 0, stream>>>(seq,   seqb,   (size_t)B_ * T_ * I_);
  k_init<<<512, 256, 0, stream>>>(h32, h16, bar);

  gru_persistent<<<NBLK, 256, 0, stream>>>(b_ih, b_hh, b_out, out,
                                           Whh_b, Wih_b, Wout_b, seqb,
                                           h32, h16, bar);
}
